// DenoiserNet_22351009808411
// MI455X (gfx1250) — matrix operand order repaired and checked by reading, not compiled, not run on hardware
//
#include <hip/hip_runtime.h>
#include <math.h>

// ---------------- model dims ----------------
#define BB   64
#define SS   100
#define EE   256
#define HH   256
#define G4   1024          // 4*H
#define KK2  512           // E + H (combined K for x||h)
#define NCAT 18

typedef __attribute__((ext_vector_type(16))) _Float16 v16h;
typedef __attribute__((ext_vector_type(8)))  _Float16 v8h;
typedef __attribute__((ext_vector_type(8)))  float    v8f;
typedef __attribute__((ext_vector_type(4)))  int      v4i;

#if __has_builtin(__builtin_amdgcn_global_load_async_to_lds_b128) && \
    __has_builtin(__builtin_amdgcn_s_wait_asynccnt)
#define HAVE_ASYNC_LDS 1
#else
#define HAVE_ASYNC_LDS 0
#endif

// ---------------- workspace layout (bytes) ----------------
static constexpr size_t OFF_W16   = 0;
static constexpr size_t SZ_W16    = (size_t)2 * 2 * G4 * KK2 * 2;          // 4 MB
static constexpr size_t OFF_BSUM  = OFF_W16 + SZ_W16;
static constexpr size_t SZ_BSUM   = (size_t)2 * 2 * G4 * 4;                // 16 KB
static constexpr size_t OFF_EMB16 = OFF_BSUM + SZ_BSUM;
static constexpr size_t SZ_EMB16  = (size_t)2 * SS * BB * EE * 2;          // 6.25 MB
static constexpr size_t OFF_Y0    = OFF_EMB16 + SZ_EMB16;
static constexpr size_t SZ_Y0     = (size_t)2 * SS * BB * HH * 2;          // 6.25 MB
static constexpr size_t OFF_HFIN  = OFF_Y0 + SZ_Y0;
static constexpr size_t SZ_HFIN   = (size_t)2 * BB * HH * 4;               // 128 KB
static constexpr size_t OFF_ENCRO = OFF_HFIN + SZ_HFIN;

__device__ __forceinline__ float sigf(float x) {
    return 1.0f / (1.0f + __expf(-x));
}

// ------------------------------------------------------------------
// Pack one LSTM layer: W16[n][k] = k<256 ? Wih[n][k] : Whh[n][k-256]
// bsum[n] = bih[n] + bhh[n]
// ------------------------------------------------------------------
__global__ void pack_weights_kernel(const float* __restrict__ Wih,
                                    const float* __restrict__ Whh,
                                    const float* __restrict__ bih,
                                    const float* __restrict__ bhh,
                                    _Float16* __restrict__ dstW,
                                    float* __restrict__ dstB) {
    int idx = blockIdx.x * blockDim.x + threadIdx.x;   // 0 .. 1024*512-1
    int n = idx >> 9;          // /512
    int k = idx & 511;
    float v = (k < EE) ? Wih[n * EE + k] : Whh[n * HH + (k - EE)];
    dstW[idx] = (_Float16)v;
    if (idx < G4) dstB[idx] = bih[idx] + bhh[idx];
}

// ------------------------------------------------------------------
// Gather embeddings -> f16, layout [lstm][t][b][e]
// ------------------------------------------------------------------
__global__ void gather_kernel(const int* __restrict__ idx_vu,
                              const int* __restrict__ idx_vo,
                              const float* __restrict__ vid,
                              _Float16* __restrict__ emb16) {
    size_t gid = (size_t)blockIdx.x * blockDim.x + threadIdx.x;  // 2*S*B*E
    int lstm = (int)(gid / ((size_t)SS * BB * EE));
    size_t r = gid % ((size_t)SS * BB * EE);
    int t = (int)(r / (BB * EE));
    int r2 = (int)(r % (BB * EE));
    int b = r2 >> 8;           // /256
    int e = r2 & 255;
    const int* ip = lstm ? idx_vo : idx_vu;
    int v = ip[b * SS + t];
    emb16[gid] = (_Float16)vid[(size_t)v * EE + e];
}

// ------------------------------------------------------------------
// Persistent 2-layer LSTM, one block per LSTM, 32 wave32 waves.
// Wave w: mhalf = w&1 (batch rows 32*mhalf..), hc = (w>>1)*16 (h columns).
// Gate matrix D[64,1024] = [x|h][64,512] @ W16^T via v_wmma_f32_16x16x32_f16.
// Cell state c lives in registers; h recirculates through LDS (f16).
// Weights stream from L2 every step (2 MB f16 per layer, L2-resident).
// ------------------------------------------------------------------
__global__ void __launch_bounds__(1024)
lstm2_kernel(const _Float16* __restrict__ W16,
             const float* __restrict__ bsum,
             const _Float16* __restrict__ emb16,
             _Float16* __restrict__ y0,
             float* __restrict__ hfin) {
    __shared__ __align__(32) _Float16 xbuf[BB * EE];   // current layer input x_t
    __shared__ __align__(32) _Float16 hbuf[BB * HH];   // recurrent h (f16)

    const int lstm  = blockIdx.x;            // 0..1
    const int tid   = threadIdx.x;
    const int lane  = tid & 31;
    const int wave  = tid >> 5;              // 0..31
    const int lanelo = lane & 15;
    const int lanehi = lane >> 4;
    const int mhalf = wave & 1;              // batch half
    const int hc    = (wave >> 1) << 4;      // h column base (0..240)
    const int mrow0 = mhalf << 5;            // 0 or 32

    const _Float16* embl  = emb16 + (size_t)lstm * SS * BB * EE;
    _Float16*       y0l   = y0    + (size_t)lstm * SS * BB * HH;
    float*          hfinl = hfin  + (size_t)lstm * BB * HH;

    for (int layer = 0; layer < 2; ++layer) {
        const _Float16* Wl    = W16  + (size_t)(lstm * 2 + layer) * G4 * KK2;
        const float*    bsuml = bsum + (size_t)(lstm * 2 + layer) * G4;
        const _Float16* xsrc  = (layer == 0) ? embl : y0l;

        // per-lane biases for this wave's i/f/g/o columns
        const float bias_i = bsuml[0 * HH + hc + lanelo];
        const float bias_f = bsuml[1 * HH + hc + lanelo];
        const float bias_g = bsuml[2 * HH + hc + lanelo];
        const float bias_o = bsuml[3 * HH + hc + lanelo];

        // zero recurrent h in LDS; zero c in registers
        {
            v8h z = {};
            v8h* hq = (v8h*)hbuf;            // 2048 v8h chunks
            hq[tid * 2 + 0] = z;
            hq[tid * 2 + 1] = z;
        }
        float cstate[2][8];
        #pragma unroll
        for (int mi = 0; mi < 2; ++mi)
            #pragma unroll
            for (int r = 0; r < 8; ++r) cstate[mi][r] = 0.0f;
        __syncthreads();

        #pragma unroll 1
        for (int t = 0; t < SS; ++t) {
            // ---- stage x_t into LDS (64x256 f16, 32B per thread) ----
            {
                const _Float16* gsrc = xsrc + (size_t)t * BB * EE + tid * 16;
#if HAVE_ASYNC_LDS
                // CDNA5 async copy: global -> LDS without VGPR round-trip,
                // tracked by ASYNCcnt. This toolchain declares the builtin
                // with generic int4-vector pointers.
                _Float16* ldst = xbuf + tid * 16;
                __builtin_amdgcn_global_load_async_to_lds_b128(
                    (v4i*)gsrc, (v4i*)ldst, 0, 0);
                __builtin_amdgcn_global_load_async_to_lds_b128(
                    (v4i*)(gsrc + 8), (v4i*)(ldst + 8), 0, 0);
                __builtin_amdgcn_s_wait_asynccnt(0);
#else
                const v8h* s = (const v8h*)gsrc;
                v8h* d = (v8h*)(xbuf + tid * 16);
                d[0] = s[0];
                d[1] = s[1];
#endif
            }
            __syncthreads();

            // ---- gates = [x|h] @ W^T  (8 WMMA tiles per wave) ----
            v8f acc[2][4];
            #pragma unroll
            for (int mi = 0; mi < 2; ++mi)
                #pragma unroll
                for (int gi = 0; gi < 4; ++gi) { v8f z = {}; acc[mi][gi] = z; }

            // K = 512, 32 per iteration. Partial unroll only: full unroll
            // makes the compiler keep all 64 weight fragments live and
            // spill them to scratch.
            #pragma unroll 2
            for (int kk = 0; kk < 16; ++kk) {
                const _Float16* abase = (kk < 8) ? xbuf : hbuf;
                const int klo = ((kk & 7) << 5) + (lanehi ? 16 : 0);
                v16h afrag[2];
                #pragma unroll
                for (int mi = 0; mi < 2; ++mi) {
                    const int row = mrow0 + (mi << 4) + lanelo;
                    const _Float16* ap = abase + row * 256 + klo;
                    v8h lo = *(const v8h*)ap;           // K: klo .. klo+7
                    v8h hi = *(const v8h*)(ap + 8);     // K: klo+8 .. klo+15
                    afrag[mi] = __builtin_shufflevector(lo, hi,
                        0, 1, 2, 3, 4, 5, 6, 7, 8, 9, 10, 11, 12, 13, 14, 15);
                }
                const int kb2 = (kk << 5) + (lanehi ? 16 : 0);
                #pragma unroll
                for (int gi = 0; gi < 4; ++gi) {
                    const int n = gi * HH + hc + lanelo;
                    v16h bfrag = *(const v16h*)(Wl + (size_t)n * KK2 + kb2);
                    #pragma unroll
                    for (int mi = 0; mi < 2; ++mi) {
                        acc[mi][gi] = __builtin_amdgcn_wmma_f32_16x16x32_f16(
                            false, afrag[mi], false, bfrag,
                            (short)0, acc[mi][gi], false, false);
                    }
                }
            }
            __syncthreads();   // all reads of xbuf/hbuf done before h rewrite

            // ---- cell update entirely in registers; write h to LDS ----
            #pragma unroll
            for (int mi = 0; mi < 2; ++mi) {
                #pragma unroll
                for (int r = 0; r < 8; ++r) {
                    const int b = mrow0 + (mi << 4) + r + (lanehi ? 8 : 0);
                    float gi_ = acc[mi][0][r] + bias_i;
                    float gf_ = acc[mi][1][r] + bias_f;
                    float gg_ = acc[mi][2][r] + bias_g;
                    float go_ = acc[mi][3][r] + bias_o;
                    float cc = sigf(gf_) * cstate[mi][r] + sigf(gi_) * tanhf(gg_);
                    cstate[mi][r] = cc;
                    float hh = sigf(go_) * tanhf(cc);
                    hbuf[b * HH + hc + lanelo] = (_Float16)hh;
                    if (layer == 0) {
                        y0l[((size_t)t * BB + b) * HH + hc + lanelo] = (_Float16)hh;
                    } else if (t == SS - 1) {
                        hfinl[b * HH + hc + lanelo] = hh;
                    }
                }
            }
            __syncthreads();   // h writes visible before next step's reads
        }
        __syncthreads();
    }
}

// ------------------------------------------------------------------
// enc_ro = relu(label_ro @ ro_W.T + ro_b)   ([64,18]x[18,256] — tiny K)
// ------------------------------------------------------------------
__global__ void encro_kernel(const float* __restrict__ label_ro,
                             const float* __restrict__ ro_W,
                             const float* __restrict__ ro_b,
                             float* __restrict__ encro) {
    int gid = blockIdx.x * blockDim.x + threadIdx.x;   // 64*256
    int b = gid >> 8;
    int h = gid & 255;
    float s = ro_b[h];
    #pragma unroll
    for (int j = 0; j < NCAT; ++j)
        s += label_ro[b * NCAT + j] * ro_W[h * NCAT + j];
    encro[gid] = fmaxf(s, 0.0f);
}

// ------------------------------------------------------------------
// dec = [h_vu | h_vo | enc_ro] @ ru_W.T + ru_b; log_softmax; KL outputs.
// One thread per batch row; out[0..63]=neg_logits, out[64]=avg_kl.
// ------------------------------------------------------------------
__global__ void head_kernel(const float* __restrict__ h_vu,
                            const float* __restrict__ h_vo,
                            const float* __restrict__ encro,
                            const float* __restrict__ ru_W,
                            const float* __restrict__ ru_b,
                            const float* __restrict__ label_ru,
                            float* __restrict__ out) {
    __shared__ float kls[BB];
    const int b = threadIdx.x;                 // 0..63
    float dec[NCAT];
    #pragma unroll
    for (int k = 0; k < NCAT; ++k) {
        float s = ru_b[k];
        const float* w = ru_W + k * (3 * HH);
        for (int j = 0; j < HH; ++j) s += h_vu[b * HH + j]  * w[j];
        for (int j = 0; j < HH; ++j) s += h_vo[b * HH + j]  * w[HH + j];
        for (int j = 0; j < HH; ++j) s += encro[b * HH + j] * w[2 * HH + j];
        dec[k] = s;
    }
    float m = dec[0];
    #pragma unroll
    for (int k = 1; k < NCAT; ++k) m = fmaxf(m, dec[k]);
    float se = 0.0f;
    #pragma unroll
    for (int k = 0; k < NCAT; ++k) se += __expf(dec[k] - m);
    float lse = m + __logf(se);

    const float inv_ln2 = 1.4426950408889634f;
    float neg = 0.0f, kl = 0.0f;
    #pragma unroll
    for (int k = 0; k < NCAT; ++k) {
        float lp = dec[k] - lse;
        float lr = label_ru[b * NCAT + k];
        neg -= lr * lp;
        kl  += lr * (__logf(lr) - lp) * inv_ln2;
    }
    out[b] = neg;
    kls[b] = kl;
    __syncthreads();
    if (b == 0) {
        float s = 0.0f;
        for (int i = 0; i < BB; ++i) s += kls[i];
        out[BB] = s / (float)BB;
    }
}

// ------------------------------------------------------------------
extern "C" void kernel_launch(void* const* d_in, const int* in_sizes, int n_in,
                              void* d_out, int out_size, void* d_ws, size_t ws_size,
                              hipStream_t stream) {
    (void)in_sizes; (void)n_in; (void)out_size; (void)ws_size;

    const int*   input_vu = (const int*)d_in[0];
    const int*   input_vo = (const int*)d_in[1];
    const float* label_ro = (const float*)d_in[2];
    const float* vid_emb  = (const float*)d_in[3];
    const float* label_ru = (const float*)d_in[4];
    // vu layer0: 5..8, vu layer1: 9..12, vo layer0: 13..16, vo layer1: 17..20
    const float* ro_W = (const float*)d_in[21];
    const float* ro_b = (const float*)d_in[22];
    const float* ru_W = (const float*)d_in[23];
    const float* ru_b = (const float*)d_in[24];

    char* ws = (char*)d_ws;
    _Float16* W16   = (_Float16*)(ws + OFF_W16);
    float*    bsum  = (float*)   (ws + OFF_BSUM);
    _Float16* emb16 = (_Float16*)(ws + OFF_EMB16);
    _Float16* y0    = (_Float16*)(ws + OFF_Y0);
    float*    hfin  = (float*)   (ws + OFF_HFIN);
    float*    encro = (float*)   (ws + OFF_ENCRO);

    // pack weights: 4 (lstm,layer) pairs, inputs start at index 5, 4 per pair
    for (int w = 0; w < 4; ++w) {
        const float* Wih = (const float*)d_in[5 + w * 4 + 0];
        const float* Whh = (const float*)d_in[5 + w * 4 + 1];
        const float* bih = (const float*)d_in[5 + w * 4 + 2];
        const float* bhh = (const float*)d_in[5 + w * 4 + 3];
        pack_weights_kernel<<<(G4 * KK2) / 256, 256, 0, stream>>>(
            Wih, Whh, bih, bhh,
            W16 + (size_t)w * G4 * KK2, bsum + (size_t)w * G4);
    }

    // gather embeddings for both LSTMs
    {
        size_t total = (size_t)2 * SS * BB * EE;
        gather_kernel<<<(unsigned)(total / 256), 256, 0, stream>>>(
            input_vu, input_vo, vid_emb, emb16);
    }

    // the two 2-layer LSTMs (one workgroup each, 32 wave32 waves)
    lstm2_kernel<<<2, 1024, 0, stream>>>(W16, bsum, emb16, y0, hfin);

    // tiny encoder
    encro_kernel<<<(BB * HH) / 256, 256, 0, stream>>>(label_ro, ro_W, ro_b, encro);

    // decoder + log-softmax + KL outputs
    head_kernel<<<1, BB, 0, stream>>>(hfin, hfin + (size_t)BB * HH, encro,
                                      ru_W, ru_b, label_ru, (float*)d_out);
}
